// MetaScaledDotProductAttetion_45810121179384
// MI455X (gfx1250) — matrix operand order repaired and checked by reading, not compiled, not run on hardware
//
#include <hip/hip_runtime.h>
#include <hip/hip_bf16.h>

// ---------------------------------------------------------------------------
// Flash-attention-2 forward, causal, B=4 H=16 S=2048 D=128, fp32 in/out.
// Matmuls via v_wmma_f32_16x16x32_f16 (f16 operands, fp32 accum).
// Block = 256 threads (8 wave32), BM=128 queries/block, BN=64 keys/tile.
// Wave-uniform causal skip keeps EXEC all-ones inside WMMA regions.
// ---------------------------------------------------------------------------

typedef __attribute__((ext_vector_type(16))) _Float16 v16h;
typedef __attribute__((ext_vector_type(8)))  _Float16 v8h;
typedef __attribute__((ext_vector_type(4)))  _Float16 h4;
typedef __attribute__((ext_vector_type(2)))  _Float16 h2;
typedef __attribute__((ext_vector_type(8)))  float    v8f;
typedef __attribute__((ext_vector_type(4)))  float    f4;

#define SEQ   2048
#define DH    128
#define BM    128     // queries per block
#define BN    64      // keys per tile
#define NWAVE 8
#define NTHR  (NWAVE * 32)

__device__ __forceinline__ v8f zero8() {
  v8f z;
#pragma unroll
  for (int i = 0; i < 8; ++i) z[i] = 0.0f;
  return z;
}

__device__ __forceinline__ v16h cat8(v8h a, v8h b) {
  return __builtin_shufflevector(a, b, 0, 1, 2, 3, 4, 5, 6, 7,
                                       8, 9, 10, 11, 12, 13, 14, 15);
}

__global__ __launch_bounds__(NTHR)
void fattn_causal_wmma_kernel(const float* __restrict__ Qg,
                              const float* __restrict__ Kg,
                              const float* __restrict__ Vg,
                              float* __restrict__ Og) {
  // Row pads keep ds_load_b128 16B-aligned (stride 144B / 272B) and spread banks.
  __shared__ _Float16 Kt[BN][DH + 8];            // 64 x 136 halves (K row-major)
  __shared__ _Float16 Vt[DH][BN + 8];            // 128 x 72 halves (V^T)
  __shared__ _Float16 Ps[NWAVE][16][BN + 8];     // per-wave P transpose scratch

  const int tid  = threadIdx.x;
  const int wave = tid >> 5;
  const int lane = tid & 31;
  const int hi   = lane >> 4;    // half-wave select (WMMA lane layout)
  const int nl   = lane & 15;

  const int q0 = blockIdx.x * BM;
  const long bh = blockIdx.y;
  const float* Qb = Qg + bh * (long)SEQ * DH;
  const float* Kb = Kg + bh * (long)SEQ * DH;
  const float* Vb = Vg + bh * (long)SEQ * DH;
  float*       Ob = Og + bh * (long)SEQ * DH;

  // 1/sqrt(128) * log2(e): softmax in exp2 domain
  const float kscale = 0.088388347648318447f * 1.4426950408889634f;

  // ---- Q tile for this wave (16 rows x 128) -> f16 A-matrix registers.
  // A 16x32 f16 layout: lane holds row (lane&15); contiguous 8-half chunks at
  // k = 32c + 8*hi and k = 32c + 8*hi + 16  (ISA 7.12.2).
  const int qrow = q0 + wave * 16 + nl;
  v16h aq[4];
#pragma unroll
  for (int c = 0; c < 4; ++c) {
    const int d0 = 32 * c + 8 * hi;
    const float* qp = Qb + (long)qrow * DH + d0;
    f4 f0 = *(const f4*)(qp + 0);
    f4 f1 = *(const f4*)(qp + 4);
    f4 f2 = *(const f4*)(qp + 16);
    f4 f3 = *(const f4*)(qp + 20);
    v16h a;
#pragma unroll
    for (int i = 0; i < 4; ++i) {
      a[i]      = (_Float16)f0[i];
      a[4 + i]  = (_Float16)f1[i];
      a[8 + i]  = (_Float16)f2[i];
      a[12 + i] = (_Float16)f3[i];
    }
    aq[c] = a;
  }

  // ---- accumulators / softmax state (8 rows per lane, C-layout)
  v8f o[8];
#pragma unroll
  for (int t = 0; t < 8; ++t) o[t] = zero8();
  float m_r[8], l_r[8];
#pragma unroll
  for (int r = 0; r < 8; ++r) { m_r[r] = -1e30f; l_r[r] = 0.0f; }

  const int jmax = q0 / BN;              // last key tile loaded (keys q0..q0+63)
  const int wave_qmax = q0 + wave * 16 + 15;

  for (int j = 0; j <= jmax; ++j) {
    const int k0 = j * BN;
    __syncthreads();   // previous iteration's Vt readers are done

    // Prefetch next tile while staging this one (global_prefetch_b8).
    if (j < jmax) {
      __builtin_prefetch(Kb + (long)(k0 + BN + (tid & 63)) * DH, 0, 1);
      __builtin_prefetch(Vb + (long)(k0 + BN + (tid & 63)) * DH, 0, 1);
    }

    // K tile: 64x128 f32 -> f16 row-major LDS (coalesced b128 reads).
#pragma unroll
    for (int i = 0; i < 8; ++i) {
      const int idx = tid + NTHR * i;    // 2048 float4 total
      const int row = idx >> 5;
      const int c4  = idx & 31;
      f4 f = *(const f4*)(Kb + (long)(k0 + row) * DH + (c4 << 2));
      h4 h;
#pragma unroll
      for (int e = 0; e < 4; ++e) h[e] = (_Float16)f[e];
      *(h4*)&Kt[row][c4 << 2] = h;
    }

    // V tile: 64x128 f32 -> transposed f16 Vt[d][k].
    // b128 reads of 4 d-values from two adjacent key rows; packed f16 pairs
    // along k stored as b32 into the transposed layout.
#pragma unroll
    for (int i = 0; i < 4; ++i) {
      const int item = tid + NTHR * i;   // 1024 (key-pair, d-quad) items
      const int d4 = item & 31;
      const int kp = item >> 5;
      const float* vp0 = Vb + (long)(k0 + 2 * kp) * DH + (d4 << 2);
      f4 a = *(const f4*)vp0;
      f4 b = *(const f4*)(vp0 + DH);
#pragma unroll
      for (int e = 0; e < 4; ++e) {
        h2 h; h[0] = (_Float16)a[e]; h[1] = (_Float16)b[e];
        *(h2*)&Vt[(d4 << 2) + e][2 * kp] = h;
      }
    }
    __syncthreads();

    // Wave-uniform causal skip: tile fully above this wave's diagonal band.
    const bool active = (k0 <= wave_qmax);

    if (active) {
      // ---- S = Q * K^T  (4 score tiles 16x16, K-dim 128 -> 4 WMMA each)
      v8f s[4];
#pragma unroll
      for (int t = 0; t < 4; ++t) {
        v8f acc = zero8();
#pragma unroll
        for (int c = 0; c < 4; ++c) {
          // B 32x16 f16 layout: lane = column n = key (t*16+nl);
          // holds k-dim d = 32c + 8*hi + [0..7] and 32c + 8*hi + 16 + [0..7] in Kt row.
          const _Float16* kr = &Kt[t * 16 + nl][32 * c + 8 * hi];
          v16h b = cat8(*(const v8h*)kr, *(const v8h*)(kr + 16));
          acc = __builtin_amdgcn_wmma_f32_16x16x32_f16(
              false, aq[c], false, b, (short)0, acc, false, false);
        }
        s[t] = acc;
      }

      // ---- scale + causal mask (branch-free) + online softmax
#pragma unroll
      for (int t = 0; t < 4; ++t) {
#pragma unroll
        for (int r = 0; r < 8; ++r) {
          const int qg = q0 + wave * 16 + r + 8 * hi;
          const int kg = k0 + t * 16 + nl;
          const float v = s[t][r] * kscale;
          s[t][r] = (kg <= qg) ? v : -1e30f;
        }
      }

      float mx[8];
#pragma unroll
      for (int r = 0; r < 8; ++r) {
        float v = fmaxf(fmaxf(s[0][r], s[1][r]), fmaxf(s[2][r], s[3][r]));
        v = fmaxf(v, __shfl_xor(v, 1));
        v = fmaxf(v, __shfl_xor(v, 2));
        v = fmaxf(v, __shfl_xor(v, 4));
        v = fmaxf(v, __shfl_xor(v, 8));   // reduce across the 16-lane n-group
        mx[r] = fmaxf(m_r[r], v);
      }

#pragma unroll
      for (int r = 0; r < 8; ++r) {
        const float alpha = exp2f(m_r[r] - mx[r]);
        float sum = 0.0f;
#pragma unroll
        for (int t = 0; t < 4; ++t) {
          const float p = exp2f(s[t][r] - mx[r]);
          s[t][r] = p;
          sum += p;
        }
        sum += __shfl_xor(sum, 1);
        sum += __shfl_xor(sum, 2);
        sum += __shfl_xor(sum, 4);
        sum += __shfl_xor(sum, 8);
        l_r[r] = l_r[r] * alpha + sum;
        m_r[r] = mx[r];
#pragma unroll
        for (int u = 0; u < 8; ++u) o[u][r] = o[u][r] * alpha;
      }

      // ---- transpose P (C-layout -> A-layout) via wave-private LDS
#pragma unroll
      for (int t = 0; t < 4; ++t)
#pragma unroll
        for (int r = 0; r < 8; ++r)
          Ps[wave][r + 8 * hi][t * 16 + nl] = (_Float16)s[t][r];
    }
    __syncthreads();   // uniform across all waves

    if (active) {
      v16h ap[2];
#pragma unroll
      for (int c = 0; c < 2; ++c) {
        const _Float16* pr = &Ps[wave][nl][32 * c + 8 * hi];
        ap[c] = cat8(*(const v8h*)pr, *(const v8h*)(pr + 16));
      }

      // ---- O += P * V  (8 output tiles 16x16, K-dim 64 -> 2 WMMA each)
#pragma unroll
      for (int t = 0; t < 8; ++t) {
        v8f acc = o[t];
#pragma unroll
        for (int c = 0; c < 2; ++c) {
          // B 32x16: lane = column n = out-dim (t*16+nl);
          // holds k = 32c + 8*hi + [0..7] and 32c + 8*hi + 16 + [0..7] in Vt (= V^T) row.
          const _Float16* vr = &Vt[t * 16 + nl][32 * c + 8 * hi];
          v16h b = cat8(*(const v8h*)vr, *(const v8h*)(vr + 16));
          acc = __builtin_amdgcn_wmma_f32_16x16x32_f16(
              false, ap[c], false, b, (short)0, acc, false, false);
        }
        o[t] = acc;
      }
    }
  }

  // ---- normalize and store (fp32 output, coalesced within 16-lane groups)
#pragma unroll
  for (int r = 0; r < 8; ++r) {
    const float inv = 1.0f / l_r[r];
    const int row = q0 + wave * 16 + r + 8 * hi;
    float* op = Ob + (long)row * DH;
#pragma unroll
    for (int t = 0; t < 8; ++t)
      op[t * 16 + nl] = o[t][r] * inv;
  }
}

extern "C" void kernel_launch(void* const* d_in, const int* in_sizes, int n_in,
                              void* d_out, int out_size, void* d_ws, size_t ws_size,
                              hipStream_t stream) {
  (void)in_sizes; (void)n_in; (void)d_ws; (void)ws_size; (void)out_size;
  const float* Q = (const float*)d_in[0];
  const float* K = (const float*)d_in[1];
  const float* V = (const float*)d_in[2];
  // d_in[3] is the boolean causal mask; causality is computed analytically.
  float* O = (float*)d_out;

  dim3 grid(SEQ / BM, 4 * 16);   // (q-tiles, B*H)
  dim3 block(NTHR);              // 8 wave32
  fattn_causal_wmma_kernel<<<grid, block, 0, stream>>>(Q, K, V, O);
}
